// GraphAttentionLayer_39805756900106
// MI455X (gfx1250) — matrix operand order repaired and checked by reading, not compiled, not run on hardware
//
#include <hip/hip_runtime.h>
#include <hip/hip_bf16.h>

typedef __attribute__((ext_vector_type(16))) __bf16 v16bf;
typedef __attribute__((ext_vector_type(8)))  __bf16 v8bf;
typedef __attribute__((ext_vector_type(2)))  __bf16 v2bf;
typedef __attribute__((ext_vector_type(8)))  float  v8f;

#define NN    8192
#define FIN   256
#define FOUT  128
#define ALPHA 0.2f

// ---------------------------------------------------------------------------
// Kernel 1: Wh = h @ W (fp32), s1 = Wh@a1, s2 = Wh@a2, and write Wh as bf16
// pre-swizzled into WMMA B-fragment layout:
//   whb[((t*8 + n)*32 + lane)*16 + e] = bf16( Wh[32t + 16*(lane>>4) + e][16n + (lane&15)] )
// which uses contiguous K order (lane: N=l%16; K = 2v+p + 16*(l/16)), same as the A gather.
// ---------------------------------------------------------------------------
__global__ __launch_bounds__(256) void k_prep(const float* __restrict__ h,
                                              const float* __restrict__ W,
                                              const float* __restrict__ a,
                                              float* __restrict__ s1,
                                              float* __restrict__ s2,
                                              __bf16* __restrict__ whb)
{
    __shared__ __align__(16) float WhLds[16][FOUT];
    const int tid  = threadIdx.x;
    const int r    = tid >> 4;        // row within 16-row tile
    const int c0   = tid & 15;        // base output column
    const int row0 = blockIdx.x * 16;

    float acc[8];
#pragma unroll
    for (int i = 0; i < 8; ++i) acc[i] = 0.f;

    const float* hrow = h + (size_t)(row0 + r) * FIN;
    for (int k = 0; k < FIN; ++k) {
        const float  hv   = hrow[k];
        const float* wrow = W + (size_t)k * FOUT + c0;
#pragma unroll
        for (int i = 0; i < 8; ++i) acc[i] = fmaf(hv, wrow[16 * i], acc[i]);
    }
#pragma unroll
    for (int i = 0; i < 8; ++i) WhLds[r][c0 + 16 * i] = acc[i];
    __syncthreads();

    // s1/s2 : per-row dot with a1=a[0:128], a2=a[128:256]
    float p1 = 0.f, p2 = 0.f;
#pragma unroll
    for (int i = 0; i < 8; ++i) {
        const int   c  = c0 + 16 * i;
        const float wv = WhLds[r][c];
        p1 = fmaf(wv, a[c], p1);
        p2 = fmaf(wv, a[FOUT + c], p2);
    }
#pragma unroll
    for (int msk = 8; msk >= 1; msk >>= 1) {   // reduce across the 16 lanes sharing r
        p1 += __shfl_xor(p1, msk, 32);
        p2 += __shfl_xor(p2, msk, 32);
    }
    if (c0 == 0) { s1[row0 + r] = p1; s2[row0 + r] = p2; }

    // pre-swizzled bf16 B-fragment tiles (this block fills one 16-K half-tile)
    if (tid < 128) {
        const int n   = tid >> 4;            // 0..7 : 16-col N tile
        const int l16 = tid & 15;            // lane%16 -> N within tile
        const int t   = row0 >> 5;           // 32-K tile index
        const int hl  = (row0 >> 4) & 1;     // which lane-half this K range feeds
        __bf16* dst = whb + (((size_t)(t * 8 + n) * 32) + hl * 16 + l16) * 16;
        v8bf lo, hi;
#pragma unroll
        for (int e = 0; e < 8; ++e) {
            lo[e] = (__bf16)WhLds[e][n * 16 + l16];
            hi[e] = (__bf16)WhLds[e + 8][n * 16 + l16];
        }
        *(v8bf*)(dst)     = lo;   // 2 coalesced b128 stores per thread
        *(v8bf*)(dst + 8) = hi;
    }
}

// ---------------------------------------------------------------------------
// Kernel 2: S2max = max(s2)
// ---------------------------------------------------------------------------
__global__ __launch_bounds__(256) void k_s2max(const float* __restrict__ s2,
                                               float* __restrict__ s2max)
{
    __shared__ float red[256];
    float m = -3.4e38f;
    for (int i = threadIdx.x; i < NN; i += 256) m = fmaxf(m, s2[i]);
    red[threadIdx.x] = m;
    __syncthreads();
    for (int off = 128; off; off >>= 1) {
        if (threadIdx.x < off)
            red[threadIdx.x] = fmaxf(red[threadIdx.x], red[threadIdx.x + off]);
        __syncthreads();
    }
    if (threadIdx.x == 0) *s2max = red[0];
}

// ---------------------------------------------------------------------------
// Kernel 3: fused masked-softmax attention, single pass over adj.
// Per block: 16 rows, 64 columns per iteration (two 32-K WMMA tiles per
// barrier pair). Shift M_i = lrelu(s1_i + max s2) makes softmax exact in one
// pass (shift invariance); exp range is bounded so no overflow/underflow.
// 8 waves; wave w owns output columns [16w,16w+16) via v_wmma_f32_16x16x32_bf16.
// ---------------------------------------------------------------------------
__global__ __launch_bounds__(256) void k_attn(const int* __restrict__ adj,
                                              const float* __restrict__ s1,
                                              const float* __restrict__ s2,
                                              const float* __restrict__ s2maxp,
                                              const __bf16* __restrict__ whb,
                                              float* __restrict__ out)
{
    __shared__ __align__(16) __bf16 Pt[16 * 64];   // P tile, row-major [m][k], 2KB
    __shared__ float rowSum[16];
    __shared__ float rowM[16];

    const int tid  = threadIdx.x;
    const int wave = tid >> 5;
    const int lane = tid & 31;
    const int row0 = blockIdx.x * 16;
    const float S2max = *s2maxp;

    if (tid < 16) {
        const float e = s1[row0 + tid] + S2max;
        rowM[tid] = fmaxf(e, ALPHA * e);           // valid shift >= true row max
    }
    __syncthreads();

    // P-tile fill: thread handles rows (wave, wave+8), cols (2*lane, 2*lane+1)
    const int   prow = wave;
    const int   pc   = 2 * lane;
    const float s1a  = s1[row0 + prow];
    const float s1b  = s1[row0 + prow + 8];
    const float Ma   = rowM[prow];
    const float Mb   = rowM[prow + 8];
    const int* adjA  = adj + (size_t)(row0 + prow) * NN;
    const int* adjB  = adj + (size_t)(row0 + prow + 8) * NN;

    float sumA = 0.f, sumB = 0.f;
    v8f c = {0.f, 0.f, 0.f, 0.f, 0.f, 0.f, 0.f, 0.f};

    // A-fragment gather addresses (contiguous K order, as in whb), per 32-K subtile:
    // lane<16: K0-15 of row m; lane>=16: K16-31.
    const int m  = lane & 15;
    const int hl = lane >> 4;
    const __bf16* a0Lo = &Pt[m * 64 + hl * 16];       // K-subtile 0
    const __bf16* a1Lo = a0Lo + 32;                   // K-subtile 1
    // B-fragment base: lane's 16 contiguous bf16 in the pre-swizzled panel
    const __bf16* bptr0 = whb + ((size_t)wave * 32 + lane) * 16;

    for (int jc = 0; jc < NN; jc += 64) {
        const int2   av2 = *(const int2*)(adjA + jc + pc);
        const int2   bv2 = *(const int2*)(adjB + jc + pc);
        const float2 s2v = *(const float2*)(s2 + jc + pc);
        __builtin_prefetch(adjA + jc + pc + 512, 0, 3);   // ~2KB ahead per row
        __builtin_prefetch(adjB + jc + pc + 512, 0, 3);

        float ea0 = s1a + s2v.x; ea0 = fmaxf(ea0, ALPHA * ea0);   // leaky relu
        float ea1 = s1a + s2v.y; ea1 = fmaxf(ea1, ALPHA * ea1);
        float eb0 = s1b + s2v.x; eb0 = fmaxf(eb0, ALPHA * eb0);
        float eb1 = s1b + s2v.y; eb1 = fmaxf(eb1, ALPHA * eb1);
        const float pa0 = (av2.x > 0) ? __expf(ea0 - Ma) : 0.f;
        const float pa1 = (av2.y > 0) ? __expf(ea1 - Ma) : 0.f;
        const float pb0 = (bv2.x > 0) ? __expf(eb0 - Mb) : 0.f;
        const float pb1 = (bv2.y > 0) ? __expf(eb1 - Mb) : 0.f;

        v2bf pa16; pa16[0] = (__bf16)pa0; pa16[1] = (__bf16)pa1;
        v2bf pb16; pb16[0] = (__bf16)pb0; pb16[1] = (__bf16)pb1;
        sumA += (float)pa16[0] + (float)pa16[1];      // sum exactly what we multiply
        sumB += (float)pb16[0] + (float)pb16[1];
        *(v2bf*)&Pt[prow * 64 + pc]       = pa16;     // single ds_store_b32 each
        *(v2bf*)&Pt[(prow + 8) * 64 + pc] = pb16;

        // B panels are Pt-independent: issue before the barrier to overlap
        const __bf16* bp0 = bptr0 + (size_t)(jc >> 5) * 4096;     // tile jc/32
        const __bf16* bp1 = bp0 + 4096;                           // tile jc/32 + 1
        const v8bf b0l = *(const v8bf*)(bp0);
        const v8bf b0h = *(const v8bf*)(bp0 + 8);
        const v8bf b1l = *(const v8bf*)(bp1);
        const v8bf b1h = *(const v8bf*)(bp1 + 8);
        __syncthreads();

        const v8bf a0l = *(const v8bf*)(a0Lo);        // ds_load_b128 x4
        const v8bf a0h = *(const v8bf*)(a0Lo + 8);
        const v8bf a1l = *(const v8bf*)(a1Lo);
        const v8bf a1h = *(const v8bf*)(a1Lo + 8);

        v16bf A0, B0, A1, B1;
#pragma unroll
        for (int i = 0; i < 8; ++i) {
            A0[i] = a0l[i]; A0[i + 8] = a0h[i];
            B0[i] = b0l[i]; B0[i + 8] = b0h[i];
            A1[i] = a1l[i]; A1[i + 8] = a1h[i];
            B1[i] = b1l[i]; B1[i + 8] = b1h[i];
        }
        c = __builtin_amdgcn_wmma_f32_16x16x32_bf16(
                false, A0, false, B0, (short)0, c, false, false);
        c = __builtin_amdgcn_wmma_f32_16x16x32_bf16(
                false, A1, false, B1, (short)0, c, false, false);
        __syncthreads();   // protect Pt before next iteration's stores
    }

    // reduce exp-sums: all 32 lanes of wave w hold partials for rows (w, w+8)
#pragma unroll
    for (int off = 16; off >= 1; off >>= 1) {
        sumA += __shfl_xor(sumA, off, 32);
        sumB += __shfl_xor(sumB, off, 32);
    }
    if (lane == 0) { rowSum[wave] = sumA; rowSum[wave + 8] = sumB; }
    __syncthreads();

    // C layout: VGPR r, lane l: row = r + 8*(l>=16), col = l%16 (within wave's slice)
    const int ncol = wave * 16 + m;
#pragma unroll
    for (int r8 = 0; r8 < 8; ++r8) {
        const int mrow = r8 + 8 * hl;
        out[(size_t)(row0 + mrow) * FOUT + ncol] = c[r8] / rowSum[mrow];
    }
}

// ---------------------------------------------------------------------------
extern "C" void kernel_launch(void* const* d_in, const int* in_sizes, int n_in,
                              void* d_out, int out_size, void* d_ws, size_t ws_size,
                              hipStream_t stream)
{
    const float* h   = (const float*)d_in[0];
    const int*   adj = (const int*)d_in[1];
    const float* W   = (const float*)d_in[2];
    const float* a   = (const float*)d_in[3];
    float* out = (float*)d_out;

    char* ws = (char*)d_ws;
    float*  s1  = (float*)ws;                    // 32 KB
    float*  s2  = (float*)(ws + 32768);          // 32 KB
    float*  s2m = (float*)(ws + 65536);          // 256 B slot
    __bf16* whb = (__bf16*)(ws + 65536 + 256);   // 2 MB pre-swizzled B panels

    hipLaunchKernelGGL(k_prep,  dim3(NN / 16), dim3(256), 0, stream, h, W, a, s1, s2, whb);
    hipLaunchKernelGGL(k_s2max, dim3(1),       dim3(256), 0, stream, s2, s2m);
    hipLaunchKernelGGL(k_attn,  dim3(NN / 16), dim3(256), 0, stream, adj, s1, s2, s2m, whb, out);
}
